// SelfAttention_52553219834403
// MI455X (gfx1250) — matrix operand order repaired and checked by reading, not compiled, not run on hardware
//
#include <hip/hip_runtime.h>
#include <cstdint>

typedef _Float16 v16h __attribute__((ext_vector_type(16)));
typedef _Float16 v8h  __attribute__((ext_vector_type(8)));
typedef _Float16 v4h  __attribute__((ext_vector_type(4)));
typedef float    v8f  __attribute__((ext_vector_type(8)));

#define LDSK 40  // 32 halves of K + 8 halves pad; rows stay 16B aligned (80B stride)

// ---------------------------------------------------------------------------
// Stage a ROWS x 32 tile into LDS f16 (row-major, LDSK stride). 256 threads.
// f32 source: load float4, convert, ds_store (compiler-tracked, DScnt).
// f16 source: GLOBAL_LOAD_ASYNC_TO_LDS_B128 direct to LDS (ASYNCcnt) —
//             no VGPR round-trip, overlaps with WMMA issue.
// All problem dims are exact multiples of the tiles; no bounds checks needed.
// ---------------------------------------------------------------------------
template<int ROWS>
__device__ __forceinline__ void stage_tile(
    const float* __restrict__ G, int ld, int r0, int c0,
    _Float16* __restrict__ lds, int tid)
{
  constexpr int EPT = ROWS * 32 / 256;       // 16 (ROWS=128) or 8 (ROWS=64)
  const int flat = tid * EPT;
  const int r = flat >> 5;
  const int c = flat & 31;
  const float* g = G + (long)(r0 + r) * ld + (c0 + c);
  _Float16* dst = lds + r * LDSK + c;
#pragma unroll
  for (int i = 0; i < EPT / 4; ++i) {
    float4 f = *(const float4*)(g + i * 4);
    v4h h = { (_Float16)f.x, (_Float16)f.y, (_Float16)f.z, (_Float16)f.w };
    *(v4h*)(dst + i * 4) = h;
  }
}

template<int ROWS>
__device__ __forceinline__ void stage_tile(
    const _Float16* __restrict__ G, int ld, int r0, int c0,
    _Float16* __restrict__ lds, int tid)
{
  constexpr int OPS = ROWS * 32 / 8;         // 16B (8 halves) per async op
  constexpr int PER = OPS / 256;             // 2 (ROWS=128) or 1 (ROWS=64)
#pragma unroll
  for (int i = 0; i < PER; ++i) {
    const int op = tid + i * 256;
    const int r = op >> 2;                   // 4 ops per 32-half row
    const int c = (op & 3) * 8;
    const _Float16* g = G + (long)(r0 + r) * ld + (c0 + c);
    // Low 32 bits of a flat LDS pointer are the LDS byte address (ISA 10.2).
    const uint32_t ldsa = (uint32_t)(uintptr_t)(lds + r * LDSK + c);
    asm volatile("global_load_async_to_lds_b128 %0, %1, off"
                 :: "v"(ldsa), "v"(g) : "memory");
  }
}

// ---------------------------------------------------------------------------
// Generic C = A * B^T WMMA GEMM.
//   A: [M,K] TA row-major (lda), B: [N,K] TB row-major (ldb), C: [M,N] TC (ldc)
//   blockIdx.z batching: A += z*zA, B += z*zB,
//                        C += (z>>zshift)*zC1 + (z & ((1<<zshift)-1))*zC2
//   CAUSAL: multiply by scale, set n>m to -inf, fast-path fully masked tiles.
// Block: 256 threads = 8 waves (wave32). Tile BM x BN, BK=32 (one WMMA K-step).
// Waves arranged 4(M) x 2(N); each wave: 2 M-tiles x (BN/32) N-tiles.
// All fragments are loaded before the WMMA burst so the matrix ops issue
// back-to-back behind a single s_wait_dscnt.
// ---------------------------------------------------------------------------
template<typename TA, typename TB, typename TC, int BM, int BN, bool CAUSAL>
__global__ __launch_bounds__(256) void gemm_abt(
    const TA* __restrict__ A, const TB* __restrict__ B, TC* __restrict__ C,
    int M, int N, int K, int lda, int ldb, int ldc,
    long zA, long zB, long zC1, long zC2, int zshift, float scale)
{
  constexpr int BK = 32;
  constexpr int WN_TILES = BN / 32;
  constexpr bool HAS_ASYNC = (sizeof(TA) == 2) || (sizeof(TB) == 2);
  __shared__ _Float16 As[BM * LDSK];
  __shared__ _Float16 Bs[BN * LDSK];

  const int tid  = threadIdx.x;
  const int lane = tid & 31;
  const int wave = tid >> 5;
  const int m0 = blockIdx.y * BM;
  const int n0 = blockIdx.x * BN;
  const int z  = blockIdx.z;
  A += (long)z * zA;
  B += (long)z * zB;
  C += (long)(z >> zshift) * zC1 + (long)(z & ((1 << zshift) - 1)) * zC2;

  if constexpr (CAUSAL) {
    if (n0 > m0 + (BM - 1)) {             // fully above the diagonal -> all masked
      const float ninf = -__builtin_inff();
      for (int i = tid; i < BM * BN; i += 256) {
        int r = i / BN, c = i % BN;
        C[(long)(m0 + r) * ldc + (n0 + c)] = (TC)ninf;
      }
      return;
    }
  }

  const int wm0 = (wave & 3) * 32;          // wave row block within tile
  const int wn0 = (wave >> 2) * (BN / 2);   // wave col block within tile

  v8f acc[2][WN_TILES] = {};

  for (int k0 = 0; k0 < K; k0 += BK) {
    __syncthreads();
    stage_tile<BM>(A, lda, m0, k0, As, tid);
    stage_tile<BN>(B, ldb, n0, k0, Bs, tid);
    if constexpr (HAS_ASYNC) asm volatile("s_wait_asynccnt 0x0" ::: "memory");
    __syncthreads();

    // A fragments: 16x32 (MxK). lane holds m = lane%16; halves 0..7 ->
    // k = kb..kb+7, halves 8..15 -> k = kb+16..+23, kb = (lane/16)*8.
    v16h af[2];
    const int ar = wm0 + (lane & 15);
    const int ak = (lane >> 4) * 8;
#pragma unroll
    for (int i = 0; i < 2; ++i) {
      const _Float16* p = &As[(ar + i * 16) * LDSK + ak];
      v8h lo = *(const v8h*)p;
      v8h hi = *(const v8h*)(p + 16);
      af[i] = __builtin_shufflevector(lo, hi, 0,1,2,3,4,5,6,7,8,9,10,11,12,13,14,15);
    }

    // B fragments: 32x16 (KxN); lane holds n = lane%16, halves 0..7 -> k = kb..kb+7,
    // halves 8..15 -> k = kb+16..+23, kb = (lane/16)*8 (same order as A),
    // Bs stored as [n][k].
    v16h bf[WN_TILES];
    const int br = wn0 + (lane & 15);
    const int bk = (lane >> 4) * 8;
#pragma unroll
    for (int j = 0; j < WN_TILES; ++j) {
      const _Float16* p = &Bs[(br + j * 16) * LDSK + bk];
      v8h lo = *(const v8h*)p;
      v8h hi = *(const v8h*)(p + 16);
      bf[j] = __builtin_shufflevector(lo, hi, 0,1,2,3,4,5,6,7,8,9,10,11,12,13,14,15);
    }

    // WMMA burst: all fragments resident, matrix ops issue back-to-back.
#pragma unroll
    for (int j = 0; j < WN_TILES; ++j) {
#pragma unroll
      for (int i = 0; i < 2; ++i) {
        acc[i][j] = __builtin_amdgcn_wmma_f32_16x16x32_f16(
            false, af[i], false, bf[j], (short)0, acc[i][j], false, false);
      }
    }
  }

  // Epilogue. C layout: lanes 0-15 -> N=lane, M=vgpr r; lanes 16-31 -> N=lane-16, M=8+r.
  const int cm = m0 + wm0 + (lane >> 4) * 8;
  const int cn = n0 + wn0 + (lane & 15);
#pragma unroll
  for (int i = 0; i < 2; ++i) {
#pragma unroll
    for (int j = 0; j < WN_TILES; ++j) {
#pragma unroll
      for (int r = 0; r < 8; ++r) {
        const int m = cm + i * 16 + r;
        const int n = cn + j * 16;
        float v = acc[i][j][r] * scale;
        if constexpr (CAUSAL) { if (n > m) v = -__builtin_inff(); }
        C[(long)m * ldc + n] = (TC)v;
      }
    }
  }
}

// ---------------------------------------------------------------------------
// f32 -> f16 cast (weights), 4 elements/thread.
// ---------------------------------------------------------------------------
__global__ __launch_bounds__(256) void cast_f32_to_f16(
    const float* __restrict__ src, _Float16* __restrict__ dst)
{
  const long i = (long)blockIdx.x * 256 + threadIdx.x;
  float4 f = *(const float4*)(src + i * 4);
  v4h h = { (_Float16)f.x, (_Float16)f.y, (_Float16)f.z, (_Float16)f.w };
  *(v4h*)(dst + i * 4) = h;
}

// ---------------------------------------------------------------------------
// Pre-LayerNorm, f16 output: one 256-thread block per 1024-wide row.
// ---------------------------------------------------------------------------
__global__ __launch_bounds__(256) void layernorm_rows(
    const float* __restrict__ x, const float* __restrict__ gamma,
    const float* __restrict__ beta, _Float16* __restrict__ y)
{
  __shared__ float s1[256], s2[256];
  const long row = blockIdx.x;
  const int tid = threadIdx.x;
  const float* xr = x + row * 1024;
  float4 v = *(const float4*)(xr + tid * 4);
  s1[tid] = v.x + v.y + v.z + v.w;
  s2[tid] = v.x * v.x + v.y * v.y + v.z * v.z + v.w * v.w;
  __syncthreads();
  for (int off = 128; off > 0; off >>= 1) {
    if (tid < off) { s1[tid] += s1[tid + off]; s2[tid] += s2[tid + off]; }
    __syncthreads();
  }
  const float mu   = s1[0] * (1.f / 1024.f);
  const float var  = s2[0] * (1.f / 1024.f) - mu * mu;
  const float rinv = rsqrtf(var + 1e-5f);
  const float4 g = *(const float4*)(gamma + tid * 4);
  const float4 b = *(const float4*)(beta  + tid * 4);
  v4h o = { (_Float16)((v.x - mu) * rinv * g.x + b.x),
            (_Float16)((v.y - mu) * rinv * g.y + b.y),
            (_Float16)((v.z - mu) * rinv * g.z + b.z),
            (_Float16)((v.w - mu) * rinv * g.w + b.w) };
  *(v4h*)(y + row * 1024 + tid * 4) = o;
}

// ---------------------------------------------------------------------------
// RoPE + head split, f16 outputs. qkv: [B*S, 3072] f32.
//   Q,K: [bh, s, 64];  Vt: [bh, 64, s] (transposed for PV B-operand).
// Reference rope (interleaved cos, half-rotation pairing):
//   out[2j]   = x[2j]  *cos(m_j) - x[32+j]*sin(m_j)
//   out[2j+1] = x[2j+1]*cos(m_j) + x[j]   *sin(m_j)
// ---------------------------------------------------------------------------
__global__ __launch_bounds__(256) void rope_split(
    const float* __restrict__ qkv, _Float16* __restrict__ Q,
    _Float16* __restrict__ Kf, _Float16* __restrict__ Vt)
{
  const int idx = blockIdx.x * 256 + threadIdx.x;   // 0 .. 2*16*2048*64-1
  const int d  = idx & 63;
  const int s  = (idx >> 6) & 2047;
  const int bh = idx >> 17;
  const int b  = bh >> 4;
  const int h  = bh & 15;
  const long row = ((long)b * 2048 + s) * 3072;
  const int col = h * 64 + d;
  const float qv = qkv[row + col];
  const float kv = qkv[row + 1024 + col];
  const float vv = qkv[row + 2048 + col];
  const int j = d >> 1;
  const float ang = (float)s * powf(10000.f, -(float)(2 * j) / 64.f);
  const float c = cosf(ang), sn = sinf(ang);
  float qo, ko;
  if (d & 1) {
    qo = qv * c + qkv[row +        h * 64 + j] * sn;
    ko = kv * c + qkv[row + 1024 + h * 64 + j] * sn;
  } else {
    qo = qv * c - qkv[row +        h * 64 + 32 + j] * sn;
    ko = kv * c - qkv[row + 1024 + h * 64 + 32 + j] * sn;
  }
  const long o = ((long)bh * 2048 + s) * 64 + d;
  Q[o]  = (_Float16)qo;
  Kf[o] = (_Float16)ko;
  Vt[((long)bh * 64 + d) * 2048 + s] = (_Float16)vv;
}

// ---------------------------------------------------------------------------
// In-place row softmax over 2048 columns. -inf entries -> exactly 0.
// ---------------------------------------------------------------------------
__global__ __launch_bounds__(256) void softmax_rows(float* __restrict__ probs)
{
  __shared__ float red[256];
  const long row = blockIdx.x;
  float* p = probs + row * 2048;
  const int tid = threadIdx.x;
  float loc[8];
  float mx = -__builtin_inff();
#pragma unroll
  for (int i = 0; i < 8; ++i) { loc[i] = p[tid + i * 256]; mx = fmaxf(mx, loc[i]); }
  red[tid] = mx; __syncthreads();
  for (int off = 128; off > 0; off >>= 1) {
    if (tid < off) red[tid] = fmaxf(red[tid], red[tid + off]);
    __syncthreads();
  }
  const float rmax = red[0];
  __syncthreads();
  float sum = 0.f;
#pragma unroll
  for (int i = 0; i < 8; ++i) { loc[i] = __expf(loc[i] - rmax); sum += loc[i]; }
  red[tid] = sum; __syncthreads();
  for (int off = 128; off > 0; off >>= 1) {
    if (tid < off) red[tid] += red[tid + off];
    __syncthreads();
  }
  const float inv = 1.f / red[0];
#pragma unroll
  for (int i = 0; i < 8; ++i) p[tid + i * 256] = loc[i] * inv;
}

// ---------------------------------------------------------------------------
// Pipeline driver.
// d_out: out f32 [4,194,304] ++ probs f32 [134,217,728].
// Scratch: x_ln (f16) lives in the 'out' region (dead before out GEMM),
// qkv (f32) lives at the head of the 'probs' region (dead before scores
// overwrite it), f16 weights + Q/K/Vt/ctx live in d_ws (40 MB total).
// ---------------------------------------------------------------------------
extern "C" void kernel_launch(void* const* d_in, const int* in_sizes, int n_in,
                              void* d_out, int out_size, void* d_ws, size_t ws_size,
                              hipStream_t stream) {
  (void)in_sizes; (void)n_in; (void)out_size; (void)ws_size;
  const float* hidden = (const float*)d_in[0];
  const float* Wqkv   = (const float*)d_in[1];
  const float* Wout   = (const float*)d_in[2];
  const float* gamma  = (const float*)d_in[3];
  const float* beta   = (const float*)d_in[4];

  float* out   = (float*)d_out;                    // [4096,1024]
  float* probs = out + 4096L * 1024;               // [32,2048,2048]
  _Float16* xln16 = (_Float16*)d_out;              // scratch alias (LN out, 8MB)
  float* qkvf  = probs;                            // scratch alias [4096,3072] f32

  _Float16* Wq16  = (_Float16*)d_ws;               // [3072,1024]
  _Float16* Wo16  = Wq16 + 3145728;                // [1024,1024]
  _Float16* Q16   = Wo16 + 1048576;                // [32,2048,64]
  _Float16* K16   = Q16  + 4194304;
  _Float16* Vt16  = K16  + 4194304;                // [32,64,2048]
  _Float16* ctx16 = Vt16 + 4194304;                // [4096,1024]

  const float kScale = 0.125f;                     // 1/sqrt(64)

  // 0) weights -> f16 (enables async LDS staging in all GEMMs)
  cast_f32_to_f16<<<3072, 256, 0, stream>>>(Wqkv, Wq16);
  cast_f32_to_f16<<<1024, 256, 0, stream>>>(Wout, Wo16);

  // 1) pre-LayerNorm -> f16
  layernorm_rows<<<4096, 256, 0, stream>>>(hidden, gamma, beta, xln16);

  // 2) QKV projection: qkv[4096,3072] = xln @ Wqkv^T   (f16 x f16 -> f32)
  gemm_abt<_Float16, _Float16, float, 128, 128, false>
      <<<dim3(24, 32, 1), 256, 0, stream>>>(
      xln16, Wq16, qkvf, 4096, 3072, 1024, 1024, 1024, 3072,
      0L, 0L, 0L, 0L, 0, 1.0f);

  // 3) RoPE + head split (Q,K rotated; V transposed), f16
  rope_split<<<16384, 256, 0, stream>>>(qkvf, Q16, K16, Vt16);

  // 4) scores per (b,h): probs = scale * Q @ K^T, causal-masked to -inf
  gemm_abt<_Float16, _Float16, float, 128, 128, true>
      <<<dim3(16, 16, 32), 256, 0, stream>>>(
      Q16, K16, probs, 2048, 2048, 64, 64, 64, 2048,
      2048L * 64, 2048L * 64, 2048L * 2048, 0L, 0, kScale);

  // 5) row softmax (in place)
  softmax_rows<<<65536, 256, 0, stream>>>(probs);

  // 6) ctx16[b,s,h*64+d] = probs[bh] @ Vt[bh]^T  (f32 A convert-staged, f16 B async)
  gemm_abt<float, _Float16, _Float16, 128, 64, false>
      <<<dim3(1, 16, 32), 256, 0, stream>>>(
      probs, Vt16, ctx16, 2048, 64, 2048, 2048, 2048, 1024,
      2048L * 2048, 64L * 2048, 2048L * 1024, 64L, 4, 1.0f);

  // 7) out = ctx16[4096,1024] @ Wout^T
  gemm_abt<_Float16, _Float16, float, 128, 128, false>
      <<<dim3(8, 32, 1), 256, 0, stream>>>(
      ctx16, Wo16, out, 4096, 1024, 1024, 1024, 1024, 1024,
      0L, 0L, 0L, 0L, 0, 1.0f);
}
